// Model_11982958756052
// MI455X (gfx1250) — matrix operand order repaired and checked by reading, not compiled, not run on hardware
//
#include <hip/hip_runtime.h>
#include <hip/hip_bf16.h>

typedef __attribute__((ext_vector_type(16))) __bf16 v16bf;
typedef __attribute__((ext_vector_type(8)))  float  v8f;

// ---------------------------------------------------------------------------
// Repack a (128 x NT*16) f32 row-major weight matrix into bf16 WMMA B-fragment
// layout: out[((kk*NT + nt)*32 + lane)*16 + e], where for 16x16x32 bf16 WMMA
// B (32x16): lane half h holds K=8h+e (e<8) and K=16+8h+(e-8) (e>=8).
// ---------------------------------------------------------------------------
__global__ void swizzleB(const float* __restrict__ W, __bf16* __restrict__ out, int NT) {
    int idx = blockIdx.x * blockDim.x + threadIdx.x;
    int total = 4 * NT * 512;               // K=128 -> 4 k-steps, 512 elems/frag
    if (idx >= total) return;
    int e    = idx & 15;
    int lane = (idx >> 4) & 31;
    int rest = idx >> 9;                    // kk*NT + nt
    int nt   = rest % NT;
    int kk   = rest / NT;
    int n    = nt * 16 + (lane & 15);
    int half = lane >> 4;
    int k    = kk * 32 + (e < 8 ? half * 8 + e : 16 + half * 8 + (e - 8));     // B layout: table order
    out[idx] = (__bf16)W[k * (NT * 16) + n];
}

// ---------------------------------------------------------------------------
// One TreeLSTM level. 16 parents per block, 128 threads (4 waves, wave32).
// Parents are the contiguous node range [lo, hi]; children of p are
// (2p-N-1, 2p-N).  hasChildren==0 for the leaf level.
// ---------------------------------------------------------------------------
__global__ void tree_level_kernel(
    const float*  __restrict__ emb,      // (VOCAB,128) f32
    const int*    __restrict__ feat,     // (N,)
    const __bf16* __restrict__ WswIou,   // 4*24*512 bf16 fragments
    const __bf16* __restrict__ UswIou,   // 4*24*512
    const __bf16* __restrict__ WswF,     // 4*8*512
    const __bf16* __restrict__ UswF,     // 4*8*512
    const float*  __restrict__ b_iou,    // (384,)
    const float*  __restrict__ b_f,      // (128,)
    __bf16*       __restrict__ hbf,      // (N,128) bf16 state
    float*        __restrict__ cst,      // (N,128) f32 state
    int N, int lo, int hi, int hasChildren)
{
    // A fragments (bf16, ISA 16x32 A layout, 4 k-steps * 512 each):
    //   [0]=x row, [2048]=h_sum, [4096]=h[ca], [6144]=h[cb]
    __shared__ __align__(64) __bf16 sA[4 * 2048];          // 16 KB
    // f32 output staging, row stride 768: iou[0..383], xf[384..511],
    // hUa[512..639], hUb[640..767]
    __shared__ float sO[16 * 768];                          // 48 KB

    const int tid  = threadIdx.x;
    const int lane = tid & 31;
    const int wave = tid >> 5;
    const int tileBase = blockIdx.x * 16;

    // ---------------- stage A operands into LDS (fragment layout) ----------
    {
        int r  = tid >> 3;                 // 0..15 : row within tile
        int k0 = (tid & 7) * 16;           // 16 K-elements per thread
        int p  = lo + tileBase + r;
        if (p > hi) p = hi;                // clamp tail rows (stores guarded)
        const float* xrow = emb + (size_t)feat[p] * 128;
        int ca = 2 * p - N - 1;
        int cb = 2 * p - N;
        for (int k = k0; k < k0 + 16; ++k) {
            int kk  = k >> 5;
            int kin = k & 31;
            int half, e;
            if (kin < 16) { half = kin >> 3;        e = kin & 7; }
            else          { half = (kin - 16) >> 3; e = 8 + ((kin - 16) & 7); }
            int pos = kk * 512 + (r + half * 16) * 16 + e;
            sA[pos] = (__bf16)xrow[k];
            if (hasChildren) {
                float ha = (float)hbf[(size_t)ca * 128 + k];
                float hb = (float)hbf[(size_t)cb * 128 + k];
                sA[2048 + pos] = (__bf16)(ha + hb);
                sA[4096 + pos] = (__bf16)ha;
                sA[6144 + pos] = (__bf16)hb;
            }
        }
    }
    __syncthreads();

    // ---------------- WMMA sweep over output column tiles -------------------
    // tiles 0..23  : iou  = x@W_iou (+ h_sum@U_iou)
    // tiles 24..31 : xf   = x@W_f
    // tiles 32..39 : hUa  = h[ca]@U_f
    // tiles 40..47 : hUb  = h[cb]@U_f
    const int ntiles = hasChildren ? 48 : 24;
    for (int t = wave; t < ntiles; t += 4) {
        const __bf16* aBase;  const __bf16* bBase;
        const __bf16* a2 = nullptr; const __bf16* b2 = nullptr;
        int NT, outCol, passes = 1;
        if (t < 24) {
            aBase = sA;         bBase = WswIou + t * 512;        NT = 24;
            outCol = t * 16;
            if (hasChildren) { passes = 2; a2 = sA + 2048; b2 = UswIou + t * 512; }
        } else if (t < 32) {
            aBase = sA;         bBase = WswF + (t - 24) * 512;   NT = 8;
            outCol = 384 + (t - 24) * 16;
        } else if (t < 40) {
            aBase = sA + 4096;  bBase = UswF + (t - 32) * 512;   NT = 8;
            outCol = 512 + (t - 32) * 16;
        } else {
            aBase = sA + 6144;  bBase = UswF + (t - 40) * 512;   NT = 8;
            outCol = 640 + (t - 40) * 16;
        }
        v8f acc = {};
        for (int pass = 0; pass < passes; ++pass) {
            const __bf16* ab = pass ? a2 : aBase;
            const __bf16* bb = pass ? b2 : bBase;
            for (int kk = 0; kk < 4; ++kk) {
                v16bf av = *(const v16bf*)(ab + kk * 512 + lane * 16);
                v16bf bv = *(const v16bf*)(bb + (size_t)kk * NT * 512 + lane * 16);
                acc = __builtin_amdgcn_wmma_f32_16x16x32_bf16(
                        false, av, false, bv, (short)0, acc, false, false);
            }
        }
        // C/D layout: VGPR r, lane L -> M = r + 8*(L/16), N = L%16
        int m0 = 8 * (lane >> 4);
        int n  = lane & 15;
        for (int r = 0; r < 8; ++r)
            sO[(m0 + r) * 768 + outCol + n] = acc[r];
    }
    __syncthreads();

    // ---------------- gating epilogue ---------------------------------------
    {
        int r  = tid >> 3;
        int j0 = (tid & 7) * 16;
        int p  = lo + tileBase + r;
        if (p <= hi) {
            int ca = 2 * p - N - 1;
            int cb = 2 * p - N;
            for (int j = j0; j < j0 + 16; ++j) {
                float iv = sO[r * 768 +       j] + b_iou[j];
                float ov = sO[r * 768 + 128 + j] + b_iou[128 + j];
                float uv = sO[r * 768 + 256 + j] + b_iou[256 + j];
                float ig = 1.0f / (1.0f + __expf(-iv));
                float og = 1.0f / (1.0f + __expf(-ov));
                float ug = tanhf(uv);
                float csum = 0.0f;
                if (hasChildren) {
                    float xf = sO[r * 768 + 384 + j] + b_f[j];
                    float fa = 1.0f / (1.0f + __expf(-(xf + sO[r * 768 + 512 + j])));
                    float fb = 1.0f / (1.0f + __expf(-(xf + sO[r * 768 + 640 + j])));
                    csum = fa * cst[(size_t)ca * 128 + j] + fb * cst[(size_t)cb * 128 + j];
                }
                float cn = ig * ug + csum;
                float hn = og * tanhf(cn);
                cst[(size_t)p * 128 + j] = cn;
                hbf[(size_t)p * 128 + j] = (__bf16)hn;
            }
        }
    }
}

// ---------------------------------------------------------------------------
// Final combine: outer(h_a,h_b)@h_c == (h_b . h_c) * h_a, then two small FCs.
// ---------------------------------------------------------------------------
__global__ void final_kernel(const __bf16* __restrict__ hC,
                             const __bf16* __restrict__ hA,
                             const __bf16* __restrict__ hB,
                             int rootC, int rootA, int rootB,
                             const float* __restrict__ fc1_w, const float* __restrict__ fc1_b,
                             const float* __restrict__ fc2_w, const float* __restrict__ fc2_b,
                             float* __restrict__ out)
{
    __shared__ float red[128];
    __shared__ float sh[128];
    __shared__ float s1[64];
    int t = threadIdx.x;                    // 128 threads
    float ha = (float)hA[(size_t)rootA * 128 + t];
    float hb = (float)hB[(size_t)rootB * 128 + t];
    float hc = (float)hC[(size_t)rootC * 128 + t];
    red[t] = hb * hc;
    __syncthreads();
    for (int s = 64; s > 0; s >>= 1) {
        if (t < s) red[t] += red[t + s];
        __syncthreads();
    }
    sh[t] = red[0] * ha;
    __syncthreads();
    if (t < 64) {
        float a = fc1_b[t];
        for (int j = 0; j < 128; ++j) a += sh[j] * fc1_w[j * 64 + t];
        s1[t] = fmaxf(a, 0.0f);
    }
    __syncthreads();
    if (t < 3) {
        float a = fc2_b[t];
        for (int j = 0; j < 64; ++j) a += s1[j] * fc2_w[j * 3 + t];
        out[t] = fmaxf(a, 0.0f);
    }
}

// ---------------------------------------------------------------------------
extern "C" void kernel_launch(void* const* d_in, const int* in_sizes, int n_in,
                              void* d_out, int out_size, void* d_ws, size_t ws_size,
                              hipStream_t stream) {
    (void)n_in; (void)out_size; (void)ws_size;
    const int*   cf    = (const int*)  d_in[0];
    const int*   af    = (const int*)  d_in[4];
    const int*   bff   = (const int*)  d_in[8];
    const float* emb   = (const float*)d_in[12];
    const float* W_iou = (const float*)d_in[13];
    const float* b_iou = (const float*)d_in[14];
    const float* U_iou = (const float*)d_in[15];
    const float* W_f   = (const float*)d_in[16];
    const float* b_f   = (const float*)d_in[17];
    const float* U_f   = (const float*)d_in[18];
    const float* fc1_w = (const float*)d_in[19];
    const float* fc1_b = (const float*)d_in[20];
    const float* fc2_w = (const float*)d_in[21];
    const float* fc2_b = (const float*)d_in[22];
    float* out = (float*)d_out;

    const int Nc = in_sizes[0];
    const int Na = in_sizes[4];
    const int Nb = in_sizes[8];

    // --- carve workspace ---------------------------------------------------
    char* p = (char*)d_ws;
    auto carve = [&](size_t bytes) -> void* {
        void* r = (void*)p;
        p += (bytes + 255) & ~(size_t)255;
        return r;
    };
    __bf16* WswIou = (__bf16*)carve((size_t)4 * 24 * 512 * sizeof(__bf16));
    __bf16* UswIou = (__bf16*)carve((size_t)4 * 24 * 512 * sizeof(__bf16));
    __bf16* WswF   = (__bf16*)carve((size_t)4 *  8 * 512 * sizeof(__bf16));
    __bf16* UswF   = (__bf16*)carve((size_t)4 *  8 * 512 * sizeof(__bf16));
    __bf16* hC = (__bf16*)carve((size_t)Nc * 128 * sizeof(__bf16));
    float*  cC = (float*) carve((size_t)Nc * 128 * sizeof(float));
    __bf16* hA = (__bf16*)carve((size_t)Na * 128 * sizeof(__bf16));
    float*  cA = (float*) carve((size_t)Na * 128 * sizeof(float));
    __bf16* hB = (__bf16*)carve((size_t)Nb * 128 * sizeof(__bf16));
    float*  cB = (float*) carve((size_t)Nb * 128 * sizeof(float));

    // --- repack weights to WMMA B fragments --------------------------------
    {
        int tot24 = 4 * 24 * 512, tot8 = 4 * 8 * 512;
        swizzleB<<<(tot24 + 255) / 256, 256, 0, stream>>>(W_iou, WswIou, 24);
        swizzleB<<<(tot24 + 255) / 256, 256, 0, stream>>>(U_iou, UswIou, 24);
        swizzleB<<<(tot8  + 255) / 256, 256, 0, stream>>>(W_f,   WswF,   8);
        swizzleB<<<(tot8  + 255) / 256, 256, 0, stream>>>(U_f,   UswF,   8);
    }

    // --- level-by-level TreeLSTM for each tree -----------------------------
    struct TreeDesc { const int* feat; int N; __bf16* h; float* c; };
    TreeDesc trees[3] = { {cf, Nc, hC, cC}, {af, Na, hA, cA}, {bff, Nb, hB, cB} };
    for (int ti = 0; ti < 3; ++ti) {
        int N = trees[ti].N;
        int depth = 0;
        while (((1 << depth) - 1) < N) depth++;          // 2^depth - 1 == N
        for (int it = 0; it < depth; ++it) {
            int lo = N + 1 - (1 << (depth - it));
            int hi = N - (1 << (depth - 1 - it));
            int count = hi - lo + 1;
            int blocks = (count + 15) / 16;
            tree_level_kernel<<<blocks, 128, 0, stream>>>(
                emb, trees[ti].feat, WswIou, UswIou, WswF, UswF,
                b_iou, b_f, trees[ti].h, trees[ti].c,
                N, lo, hi, (it > 0) ? 1 : 0);
        }
    }

    // --- fuse + MLP head ----------------------------------------------------
    final_kernel<<<1, 128, 0, stream>>>(hC, hA, hB, Nc - 1, Na - 1, Nb - 1,
                                        fc1_w, fc1_b, fc2_w, fc2_b, out);
}
